// Spatial_Attention_88175678587419
// MI455X (gfx1250) — compile-verified, matrix operand order checked by reading, not run on hardware
//
#include <hip/hip_runtime.h>
#include <hip/hip_bf16.h>

// ---------------------------------------------------------------------------
// Spatial attention (CAM refinement) for MI455X / gfx1250, bf16 WMMA pipeline.
//   B=32, C=2048, H=W=32 (HW=1024), mid=128
// Stages:
//   0) Wq/Wk f32 -> bf16
//   1) f1 = Wq*feat, f2 = Wk*feat          (bf16 WMMA, K=2048, 16x128 tiles)
//   2) S  = f1^T * f2  (f32 out)           (bf16 WMMA, K=128)
//   3) row softmax(S) -> P (bf16, in place over S buffer, row stride 2048)
//   4) out = alpha * (cam x P) + cam       (bf16 WMMA, K=1024, 64x64 tiles)
// ---------------------------------------------------------------------------

typedef __bf16 bf16_t;
typedef __attribute__((ext_vector_type(16))) __bf16 v16bf;
typedef __attribute__((ext_vector_type(8)))  float  v8f;

#define HWN  1024
#define CCH  2048
#define MID  128
#define NB   32
#define LDT  40   // padded LDS row stride (bf16 elems): 16B-aligned rows, no bank conflicts

// 16x32 bf16 WMMA operand fragment from an LDS tile stored row-major with
// stride `ld` ("row" = M for A, = N for B staged transposed as [n][k]).
// Per-lane layout (ISA 7.12.2, 16-bit A 16x32):
//   row = lane&15 ; element e: K = (e>>3)*16 + (lane>>4)*8 + (e&7)
// Each 8-element run is 16 contiguous, aligned bytes in LDS -> ds_load_b128.
__device__ __forceinline__ v16bf frag_row16(const bf16_t* base, int lane, int ld) {
  const int r  = lane & 15;
  const int kh = (lane >> 4) << 3;
  v16bf f;
#pragma unroll
  for (int g = 0; g < 2; ++g)
#pragma unroll
    for (int w = 0; w < 8; ++w)
      f[g * 8 + w] = base[r * ld + g * 16 + kh + w];
  return f;
}

__device__ __forceinline__ v8f zero8() {
  v8f z = {0.f, 0.f, 0.f, 0.f, 0.f, 0.f, 0.f, 0.f};
  return z;
}

// Pack two floats -> two bf16 -> one aligned 32-bit DS store (idx must be even).
__device__ __forceinline__ void st_pair_f32(bf16_t* lds, int idx, float a, float b) {
  union { bf16_t h[2]; unsigned u; } p;
  p.h[0] = (bf16_t)a;
  p.h[1] = (bf16_t)b;
  *(unsigned*)(lds + idx) = p.u;
}
__device__ __forceinline__ void st_pair_bf(bf16_t* lds, int idx, bf16_t a, bf16_t b) {
  union { bf16_t h[2]; unsigned u; } p;
  p.h[0] = a;
  p.h[1] = b;
  *(unsigned*)(lds + idx) = p.u;
}

// ---------------------------------------------------------------- kernel 0 --
__global__ void __launch_bounds__(256)
k_cvt_w(const float* __restrict__ Wq, const float* __restrict__ Wk,
        bf16_t* __restrict__ Wqb, bf16_t* __restrict__ Wkb, int n) {
  int idx = blockIdx.x * 256 + threadIdx.x;
  if (idx < n) {
    Wqb[idx] = (bf16_t)Wq[idx];
    Wkb[idx] = (bf16_t)Wk[idx];
  }
}

// ---------------------------------------------------------------- kernel 1 --
// f1[b][m][n], f2[b][m][n] (bf16, ld=HWN).  Block = 4 waves, tile 16(M)x128(N);
// each wave keeps Wq/Wk A-fragments and sweeps 2 B-subtiles -> 4 WMMA per step.
__global__ void __launch_bounds__(128)
k_proj(const float* __restrict__ feat, const bf16_t* __restrict__ Wqb,
       const bf16_t* __restrict__ Wkb, bf16_t* __restrict__ f1,
       bf16_t* __restrict__ f2) {
  __shared__ bf16_t Aq[16 * LDT];
  __shared__ bf16_t Ak[16 * LDT];
  __shared__ bf16_t Bt[128 * LDT];          // feat tile, transposed [n][k]
  const int t = threadIdx.x, lane = t & 31, wave = t >> 5;
  const int n0 = blockIdx.x * 128, m0 = blockIdx.y * 16, b = blockIdx.z;
  const size_t featB = (size_t)b * CCH * HWN;
  v8f accq[2] = {zero8(), zero8()};
  v8f acck[2] = {zero8(), zero8()};

  for (int k0 = 0; k0 < CCH; k0 += 32) {
    { // weight tiles: 16x32 bf16 each, 4 contiguous elems / thread (b64 LDS ops)
      const int m = t >> 3, j = (t & 7) * 4;
#pragma unroll
      for (int q = 0; q < 4; ++q) {
        Aq[m * LDT + j + q] = Wqb[(m0 + m) * CCH + k0 + j + q];
        Ak[m * LDT + j + q] = Wkb[(m0 + m) * CCH + k0 + j + q];
      }
    }
    { // feat [32 k][128 n] f32 -> Bt[n][k] bf16; pair K rows -> packed b32 stores
      const int k  = (t & 15) * 2;
      const int nq = (t >> 4) * 16;
      const float* s0 = feat + featB + (size_t)(k0 + k) * HWN + n0 + nq;
      const float* s1 = s0 + HWN;
      if (k0 + 32 < CCH)
        __builtin_prefetch(s0 + 32 * HWN, 0, 0);
#pragma unroll
      for (int w = 0; w < 16; ++w)
        st_pair_f32(Bt, (nq + w) * LDT + k, s0[w], s1[w]);
    }
    __syncthreads();
    v16bf aq = frag_row16(Aq, lane, LDT);
    v16bf ak = frag_row16(Ak, lane, LDT);
#pragma unroll
    for (int s = 0; s < 2; ++s) {
      v16bf bfr = frag_row16(Bt + (wave * 32 + s * 16) * LDT, lane, LDT);
      accq[s] = __builtin_amdgcn_wmma_f32_16x16x32_bf16(false, aq, false, bfr,
                                                        (short)0, accq[s], false, false);
      acck[s] = __builtin_amdgcn_wmma_f32_16x16x32_bf16(false, ak, false, bfr,
                                                        (short)0, acck[s], false, false);
    }
    __syncthreads();
  }
  // D layout: N = lane&15, M = v + 8*(lane>>4)
  const int mb = m0 + ((lane >> 4) << 3);
  const size_t ob = (size_t)b * MID * HWN;
#pragma unroll
  for (int s = 0; s < 2; ++s) {
    const int nn = n0 + wave * 32 + s * 16 + (lane & 15);
#pragma unroll
    for (int v = 0; v < 8; ++v) {
      f1[ob + (size_t)(mb + v) * HWN + nn] = (bf16_t)accq[s][v];
      f2[ob + (size_t)(mb + v) * HWN + nn] = (bf16_t)acck[s][v];
    }
  }
}

// ---------------------------------------------------------------- kernel 2 --
// S[b][i][j] = sum_m f1[b][m][i] * f2[b][m][j]   (f32 out, K=128)
__global__ void __launch_bounds__(128)
k_scores(const bf16_t* __restrict__ f1, const bf16_t* __restrict__ f2,
         float* __restrict__ S) {
  __shared__ bf16_t At[16 * LDT];           // [i][m]
  __shared__ bf16_t Bt[64 * LDT];           // [j][m]
  const int t = threadIdx.x, lane = t & 31, wave = t >> 5;
  const int j0 = blockIdx.x * 64, i0 = blockIdx.y * 16, b = blockIdx.z;
  const size_t fb = (size_t)b * MID * HWN;
  v8f acc = zero8();

  for (int k0 = 0; k0 < MID; k0 += 32) {
    const int m2 = (t & 15) * 2;            // even K row pair
    { // A: At[ii][m] = f1[k0+m][i0+ii], packed b32 stores over (m, m+1)
      const int ig = t >> 4;                // 8 groups x 2 rows
#pragma unroll
      for (int q = 0; q < 2; ++q) {
        const int ii = ig * 2 + q;
        st_pair_bf(At, ii * LDT + m2,
                   f1[fb + (size_t)(k0 + m2) * HWN + i0 + ii],
                   f1[fb + (size_t)(k0 + m2 + 1) * HWN + i0 + ii]);
      }
    }
    { // B: Bt[jj][m] = f2[k0+m][j0+jj], packed b32 stores over (m, m+1)
      const int jq = (t >> 4) * 8;
      const bf16_t* r0 = f2 + fb + (size_t)(k0 + m2) * HWN + j0 + jq;
      const bf16_t* r1 = r0 + HWN;
#pragma unroll
      for (int w = 0; w < 8; ++w)
        st_pair_bf(Bt, (jq + w) * LDT + m2, r0[w], r1[w]);
    }
    __syncthreads();
    v16bf a  = frag_row16(At, lane, LDT);
    v16bf bb = frag_row16(Bt + wave * 16 * LDT, lane, LDT);
    acc = __builtin_amdgcn_wmma_f32_16x16x32_bf16(false, a, false, bb,
                                                  (short)0, acc, false, false);
    __syncthreads();
  }
  const int jj = j0 + wave * 16 + (lane & 15);
  const int ib = i0 + ((lane >> 4) << 3);
  const size_t sb = (size_t)b * HWN * HWN;
#pragma unroll
  for (int v = 0; v < 8; ++v)
    S[sb + (size_t)(ib + v) * HWN + jj] = acc[v];
}

// ---------------------------------------------------------------- kernel 3 --
// Row softmax of S, written IN PLACE as bf16 over the front of each f32 row
// (P row stride = 2*HWN bf16).  All global reads of a row happen before the
// first barrier, so the overlapping writes are race-free.
__global__ void __launch_bounds__(256)
k_softmax(float* __restrict__ S) {
  __shared__ float red[256];
  const int t = threadIdx.x, i = blockIdx.x, b = blockIdx.y;
  float* row = S + ((size_t)b * HWN + i) * HWN;
  float v[4];
  float m = -3.4e38f;
#pragma unroll
  for (int q = 0; q < 4; ++q) { v[q] = row[t + q * 256]; m = fmaxf(m, v[q]); }
  red[t] = m; __syncthreads();
  for (int s = 128; s > 0; s >>= 1) {
    if (t < s) red[t] = fmaxf(red[t], red[t + s]);
    __syncthreads();
  }
  m = red[0]; __syncthreads();
  float sum = 0.f;
#pragma unroll
  for (int q = 0; q < 4; ++q) { v[q] = __expf(v[q] - m); sum += v[q]; }
  red[t] = sum; __syncthreads();
  for (int s = 128; s > 0; s >>= 1) {
    if (t < s) red[t] += red[t + s];
    __syncthreads();
  }
  const float inv = 1.0f / red[0];
  bf16_t* prow = (bf16_t*)row;
#pragma unroll
  for (int q = 0; q < 4; ++q) prow[t + q * 256] = (bf16_t)(v[q] * inv);
}

// ---------------------------------------------------------------- kernel 4 --
// out[b][c][i] = alpha * sum_j cam[b][c][j] * P[b][i][j] + cam[b][c][i]
// Block tile 64(c) x 64(i); each wave: one B fragment, four A fragments,
// four accumulators -> 4 WMMA per B-fragment load.
__global__ void __launch_bounds__(128)
k_apply(const float* __restrict__ cam, const float* __restrict__ S,
        const float* __restrict__ alphap, float* __restrict__ out) {
  __shared__ bf16_t At[64 * LDT];           // [c][j]
  __shared__ bf16_t Bt[64 * LDT];           // [i][j]
  const int t = threadIdx.x, lane = t & 31, wave = t >> 5;
  const int i0 = blockIdx.x * 64, c0 = blockIdx.y * 64, b = blockIdx.z;
  const size_t camB = (size_t)b * CCH * HWN;
  const bf16_t* P = (const bf16_t*)(S + (size_t)b * HWN * HWN);  // row stride 2*HWN
  v8f acc[4] = {zero8(), zero8(), zero8(), zero8()};

  for (int k0 = 0; k0 < HWN; k0 += 32) {
    { // A: cam tile [64 c][32 j] f32 -> bf16; 16 contiguous elems / thread
      const int c = t >> 1, jh = (t & 1) * 16;
      const float* src = cam + camB + (size_t)(c0 + c) * HWN + k0 + jh;
      if (k0 + 32 < HWN)
        __builtin_prefetch(src + 32, 0, 0);
#pragma unroll
      for (int w = 0; w < 16; w += 2)
        st_pair_f32(At, c * LDT + jh + w, src[w], src[w + 1]);
    }
    { // B: Bt[n=i][k=j] = P[i0+n][k0+k]; 16 contiguous bf16 / thread (b128s)
      const int n = t >> 1, kh = (t & 1) * 16;
      const bf16_t* src = P + (size_t)(i0 + n) * (2 * HWN) + k0 + kh;
#pragma unroll
      for (int w = 0; w < 16; ++w)
        Bt[n * LDT + kh + w] = src[w];
    }
    __syncthreads();
    v16bf bb = frag_row16(Bt + wave * 16 * LDT, lane, LDT);
#pragma unroll
    for (int r = 0; r < 4; ++r) {
      v16bf a = frag_row16(At + r * 16 * LDT, lane, LDT);
      acc[r] = __builtin_amdgcn_wmma_f32_16x16x32_bf16(false, a, false, bb,
                                                       (short)0, acc[r], false, false);
    }
    __syncthreads();
  }
  const float alpha = alphap[0];
  const int ii = i0 + wave * 16 + (lane & 15);
#pragma unroll
  for (int r = 0; r < 4; ++r) {
    const int cb = c0 + r * 16 + ((lane >> 4) << 3);
#pragma unroll
    for (int v = 0; v < 8; ++v) {
      const size_t idx = camB + (size_t)(cb + v) * HWN + ii;
      out[idx] = alpha * acc[r][v] + cam[idx];
    }
  }
}

// -------------------------------------------------------------------- host --
extern "C" void kernel_launch(void* const* d_in, const int* in_sizes, int n_in,
                              void* d_out, int out_size, void* d_ws, size_t ws_size,
                              hipStream_t stream) {
  const float* feat  = (const float*)d_in[0];
  const float* cam   = (const float*)d_in[1];
  const float* Wq    = (const float*)d_in[2];
  const float* Wk    = (const float*)d_in[3];
  const float* alpha = (const float*)d_in[4];
  float* out = (float*)d_out;

  // workspace layout (bytes):
  //   [0,       512K)  Wq bf16
  //   [512K,    1M)    Wk bf16
  //   [1M,      9M)    f1 bf16  [B][128][1024]
  //   [9M,      17M)   f2 bf16
  //   [17M,     145M)  S  f32   [B][1024][1024]  (P bf16 reuses this in place)
  char* ws = (char*)d_ws;
  bf16_t* Wqb = (bf16_t*)(ws);
  bf16_t* Wkb = (bf16_t*)(ws + ((size_t)512 << 10));
  bf16_t* f1  = (bf16_t*)(ws + ((size_t)1 << 20));
  bf16_t* f2  = (bf16_t*)(ws + ((size_t)1 << 20) + ((size_t)8 << 20));
  float*  S   = (float* )(ws + ((size_t)1 << 20) + ((size_t)16 << 20));

  k_cvt_w<<<dim3((MID * CCH + 255) / 256), 256, 0, stream>>>(Wq, Wk, Wqb, Wkb,
                                                             MID * CCH);
  k_proj<<<dim3(HWN / 128, MID / 16, NB), 128, 0, stream>>>(feat, Wqb, Wkb, f1, f2);
  k_scores<<<dim3(HWN / 64, HWN / 16, NB), 128, 0, stream>>>(f1, f2, S);
  k_softmax<<<dim3(HWN, NB), 256, 0, stream>>>(S);
  k_apply<<<dim3(HWN / 64, CCH / 64, NB), 128, 0, stream>>>(cam, S, alpha, out);
}
